// Token_QK_Attention_36670430773400
// MI455X (gfx1250) — matrix operand order repaired and checked by reading, not compiled, not run on hardware
//
#include <hip/hip_runtime.h>

// ---------------------------------------------------------------------------
// Spiking Token-QK attention for MI455X (gfx1250, wave32, WMMA).
// Pipeline: [q,k] --(bf16 WMMA GEMM + BN + LIF)--> u8 spikes
//           q spikes --(per-head packed sum + LIF(0.5))--> u8 attn
//           (k spikes AND attn) --(bf16 WMMA GEMM + bias/BN + LIF)--> f32 out
// ---------------------------------------------------------------------------

typedef __bf16          v16bf __attribute__((ext_vector_type(16)));
typedef float           v8f   __attribute__((ext_vector_type(8)));
typedef unsigned short  u16x4 __attribute__((ext_vector_type(4)));

#define T_   4
#define B_   8
#define C_   512
#define N_   1024
#define H_   8
#define DH_  64
#define KTILES (C_ / 32)   // 16 K-chunks of 32
#define MTILES (C_ / 16)   // 32 M-tiles of 16

// ---------------------------------------------------------------------------
// Swizzle f32 weights [Cout,Cin] into the bf16 A-fragment layout for
// v_wmma_f32_16x16x32_bf16:  lane L holds row M = L%16, element e holds
// K = e + (L>=16 ? 16:0)  (contiguous, matches B staging).
// One tile = 32 lanes x 16 bf16 = 512 bf16, tiles indexed mtile*KTILES + kc.
// ---------------------------------------------------------------------------
__global__ void prep_swizzle(const float* __restrict__ w0,
                             const float* __restrict__ w1,
                             const float* __restrict__ w2,
                             unsigned short* __restrict__ dst)
{
    int id   = blockIdx.x * blockDim.x + threadIdx.x;   // 3 * 512 tiles * 32 lanes
    int mat  = id >> 14;            // / 16384
    int rem  = id & 16383;
    int tile = rem >> 5;
    int lane = rem & 31;
    int mt   = tile >> 4;           // / KTILES
    int kc   = tile & (KTILES - 1);
    const float* w = (mat == 0) ? w0 : (mat == 1) ? w1 : w2;
    int m     = mt * 16 + (lane & 15);
    int kbase = kc * 32 + ((lane >= 16) ? 16 : 0);
    v16bf out;
#pragma unroll
    for (int e = 0; e < 16; ++e) {
        int k  = kbase + e;
        out[e] = (__bf16)w[(size_t)m * C_ + k];
    }
    *(v16bf*)(dst + (size_t)mat * C_ * C_ + (size_t)tile * 512 + lane * 16) = out;
}

// Fold BatchNorm (and proj bias) into per-channel scale/shift.
__global__ void prep_scales(const float* __restrict__ qg, const float* __restrict__ qb,
                            const float* __restrict__ qm, const float* __restrict__ qv,
                            const float* __restrict__ kg, const float* __restrict__ kb,
                            const float* __restrict__ km, const float* __restrict__ kv,
                            const float* __restrict__ pg, const float* __restrict__ pb,
                            const float* __restrict__ pm, const float* __restrict__ pv,
                            const float* __restrict__ bias,
                            float* __restrict__ out)     // [3][2][512]: scale,shift
{
    int id  = blockIdx.x * blockDim.x + threadIdx.x;     // 3*512
    int mat = id >> 9;
    int d   = id & (C_ - 1);
    const float *g, *b, *m, *v;
    if (mat == 0)      { g = qg; b = qb; m = qm; v = qv; }
    else if (mat == 1) { g = kg; b = kb; m = km; v = kv; }
    else               { g = pg; b = pb; m = pm; v = pv; }
    float inv = g[d] * rsqrtf(v[d] + 1e-5f);
    float sh  = b[d] - m[d] * inv;
    if (mat == 2) sh += bias[d] * inv;
    out[mat * 1024 + d]        = inv;
    out[mat * 1024 + C_ + d]   = sh;
}

// ---------------------------------------------------------------------------
// GEMM (f32 input) + BN + multi-step LIF -> u8 spikes.
// Workgroup = 8 waves, output tile M=128, N=32, all T=4 timesteps.
// Each wave: 16(M) x 32(N) x 4(t) -> 8 f32x8 accumulators.
// B chunk (32K x 32N, 4 t) staged to LDS in B-fragment layout
// (col = lane%16, K = e + 16*(lane>=16)).
// ---------------------------------------------------------------------------
__global__ __launch_bounds__(256) void gemm_bn_lif_f32(
    const float*          __restrict__ X,     // [T,B,C,N] f32
    const unsigned short* __restrict__ Wsw,   // swizzled bf16 weights
    const float*          __restrict__ scale, // [512]
    const float*          __restrict__ shift, // [512]
    unsigned char*        __restrict__ S)     // [T,B,C,N] u8 spikes
{
    __shared__ alignas(64) unsigned short Bsh[T_][2][32][16];

    const int n0    = blockIdx.x * 32;
    const int mwg   = blockIdx.y * 128;
    const int b     = blockIdx.z;
    const int tid   = threadIdx.x;
    const int lane  = tid & 31;
    const int wave  = tid >> 5;
    const int mbase = mwg + wave * 16;
    const int mtile = mbase >> 4;

    // staging role: thread -> (n column nn, k group g covering k = 4g..4g+3)
    const int nn   = tid & 31;
    const int g    = tid >> 5;
    const int kk   = g * 4;
    const int fpos = nn >> 4;
    const int Lpos = (nn & 15) + ((kk & 16) ? 16 : 0);
    const int e0   = kk & 15;

    v8f acc[T_][2];
#pragma unroll
    for (int t = 0; t < T_; ++t)
#pragma unroll
        for (int nt = 0; nt < 2; ++nt) acc[t][nt] = v8f{};

    for (int kc = 0; kc < KTILES; ++kc) {
        const int k0 = kc * 32;
        __syncthreads();
#pragma unroll
        for (int t = 0; t < T_; ++t) {
            const float* src =
                X + ((((size_t)t * B_ + b) * C_ + (k0 + kk)) * N_ + (n0 + nn));
            u16x4 vv;
#pragma unroll
            for (int j = 0; j < 4; ++j) {
                __bf16 hb = (__bf16)src[(size_t)j * N_];
                vv[j] = __builtin_bit_cast(unsigned short, hb);
            }
            *(u16x4*)&Bsh[t][fpos][Lpos][e0] = vv;
        }
        __syncthreads();

        const v16bf a = *(const v16bf*)(
            Wsw + ((size_t)(mtile * KTILES + kc)) * 512 + lane * 16);
#pragma unroll
        for (int t = 0; t < T_; ++t)
#pragma unroll
            for (int nt = 0; nt < 2; ++nt) {
                const v16bf bf = *(const v16bf*)(&Bsh[t][nt][lane][0]);
                acc[t][nt] = __builtin_amdgcn_wmma_f32_16x16x32_bf16(
                    false, a, false, bf, (short)0, acc[t][nt], false, false);
            }
    }

    // Epilogue: BN + LIF over t, write binary spikes (u8).
    const int half = lane >> 4;
#pragma unroll
    for (int r = 0; r < 8; ++r) {
        const int d  = mbase + r + half * 8;
        const float sc = scale[d];
        const float sh = shift[d];
#pragma unroll
        for (int nt = 0; nt < 2; ++nt) {
            const int n = n0 + nt * 16 + (lane & 15);
            float v = 0.f;
#pragma unroll
            for (int t = 0; t < T_; ++t) {
                float y = acc[t][nt][r] * sc + sh;
                v = 0.5f * (v + y);                 // v += (y - v)/tau, tau=2
                float s = (v >= 1.0f) ? 1.0f : 0.0f;
                v = (s != 0.f) ? 0.f : v;           // hard reset
                S[(((size_t)t * B_ + b) * C_ + d) * N_ + n] = (unsigned char)s;
            }
        }
    }
}

// ---------------------------------------------------------------------------
// Per-head sum of q spikes over head_dim (packed u8x4 adds, max 64 < 256 so
// no cross-byte carry) then LIF with v_th = 0.5 -> attn spikes (u8).
// One thread handles 4 consecutive n.
// ---------------------------------------------------------------------------
__global__ void qsum_attn(const unsigned char* __restrict__ qs,
                          unsigned char*       __restrict__ attn)
{
    int id = blockIdx.x * blockDim.x + threadIdx.x;   // B*H*(N/4) = 16384
    int n4 = id & (N_ / 4 - 1);
    int h  = (id >> 8) & (H_ - 1);
    int b  = id >> 11;

    unsigned p[T_] = {0u, 0u, 0u, 0u};
#pragma unroll 4
    for (int dh = 0; dh < DH_; ++dh) {
        int c = h * DH_ + dh;
#pragma unroll
        for (int t = 0; t < T_; ++t)
            p[t] += *(const unsigned*)(qs +
                (((size_t)t * B_ + b) * C_ + c) * N_ + (size_t)n4 * 4);
    }
    unsigned outp[T_] = {0u, 0u, 0u, 0u};
#pragma unroll
    for (int j = 0; j < 4; ++j) {
        float v = 0.f;
#pragma unroll
        for (int t = 0; t < T_; ++t) {
            float y = (float)((p[t] >> (8 * j)) & 255u);
            v = 0.5f * (v + y);
            if (v >= 0.5f) { outp[t] |= (1u << (8 * j)); v = 0.f; }
        }
    }
#pragma unroll
    for (int t = 0; t < T_; ++t)
        *(unsigned*)(attn + (((size_t)t * B_ + b) * H_ + h) * N_ + (size_t)n4 * 4)
            = outp[t];
}

// ---------------------------------------------------------------------------
// Projection GEMM: B operand is (k_spike AND attn_spike) reconstructed
// exactly as bf16 {0,1} during LDS staging. Epilogue: bias/BN + LIF -> f32.
// ---------------------------------------------------------------------------
__global__ __launch_bounds__(256) void gemm_proj_bn_lif(
    const unsigned char*  __restrict__ KS,    // [T,B,C,N] u8
    const unsigned char*  __restrict__ ATT,   // [T,B,H,N] u8
    const unsigned short* __restrict__ Wsw,
    const float*          __restrict__ scale,
    const float*          __restrict__ shift,
    float*                __restrict__ OUT)   // [T,B,C,N] f32 spikes
{
    __shared__ alignas(64) unsigned short Bsh[T_][2][32][16];

    const int n0    = blockIdx.x * 32;
    const int mwg   = blockIdx.y * 128;
    const int b     = blockIdx.z;
    const int tid   = threadIdx.x;
    const int lane  = tid & 31;
    const int wave  = tid >> 5;
    const int mbase = mwg + wave * 16;
    const int mtile = mbase >> 4;

    const int nn   = tid & 31;
    const int g    = tid >> 5;
    const int kk   = g * 4;
    const int fpos = nn >> 4;
    const int Lpos = (nn & 15) + ((kk & 16) ? 16 : 0);
    const int e0   = kk & 15;

    v8f acc[T_][2];
#pragma unroll
    for (int t = 0; t < T_; ++t)
#pragma unroll
        for (int nt = 0; nt < 2; ++nt) acc[t][nt] = v8f{};

    for (int kc = 0; kc < KTILES; ++kc) {
        const int k0 = kc * 32;
        const int h  = (k0 + kk) >> 6;   // head of this 4-channel group
        __syncthreads();
#pragma unroll
        for (int t = 0; t < T_; ++t) {
            const unsigned char* src =
                KS + (((size_t)t * B_ + b) * C_ + (k0 + kk)) * N_ + (n0 + nn);
            const unsigned char a8 =
                ATT[(((size_t)t * B_ + b) * H_ + h) * N_ + (n0 + nn)];
            u16x4 vv;
#pragma unroll
            for (int j = 0; j < 4; ++j)
                vv[j] = (src[(size_t)j * N_] & a8) ? (unsigned short)0x3F80u
                                                   : (unsigned short)0u;
            *(u16x4*)&Bsh[t][fpos][Lpos][e0] = vv;
        }
        __syncthreads();

        const v16bf a = *(const v16bf*)(
            Wsw + ((size_t)(mtile * KTILES + kc)) * 512 + lane * 16);
#pragma unroll
        for (int t = 0; t < T_; ++t)
#pragma unroll
            for (int nt = 0; nt < 2; ++nt) {
                const v16bf bf = *(const v16bf*)(&Bsh[t][nt][lane][0]);
                acc[t][nt] = __builtin_amdgcn_wmma_f32_16x16x32_bf16(
                    false, a, false, bf, (short)0, acc[t][nt], false, false);
            }
    }

    const int half = lane >> 4;
#pragma unroll
    for (int r = 0; r < 8; ++r) {
        const int d  = mbase + r + half * 8;
        const float sc = scale[d];
        const float sh = shift[d];
#pragma unroll
        for (int nt = 0; nt < 2; ++nt) {
            const int n = n0 + nt * 16 + (lane & 15);
            float v = 0.f;
#pragma unroll
            for (int t = 0; t < T_; ++t) {
                float y = acc[t][nt][r] * sc + sh;
                v = 0.5f * (v + y);
                float s = (v >= 1.0f) ? 1.0f : 0.0f;
                v = (s != 0.f) ? 0.f : v;
                OUT[(((size_t)t * B_ + b) * C_ + d) * N_ + n] = s;
            }
        }
    }
}

// ---------------------------------------------------------------------------
extern "C" void kernel_launch(void* const* d_in, const int* in_sizes, int n_in,
                              void* d_out, int out_size, void* d_ws, size_t ws_size,
                              hipStream_t stream)
{
    (void)in_sizes; (void)n_in; (void)out_size; (void)ws_size;

    const float* q      = (const float*)d_in[0];
    const float* k      = (const float*)d_in[1];
    const float* q_w    = (const float*)d_in[2];
    const float* k_w    = (const float*)d_in[3];
    const float* proj_w = (const float*)d_in[4];
    const float* proj_b = (const float*)d_in[5];
    const float* qg = (const float*)d_in[6],  *qb = (const float*)d_in[7];
    const float* qm = (const float*)d_in[8],  *qv = (const float*)d_in[9];
    const float* kg = (const float*)d_in[10], *kb = (const float*)d_in[11];
    const float* km = (const float*)d_in[12], *kv = (const float*)d_in[13];
    const float* pg = (const float*)d_in[14], *pb = (const float*)d_in[15];
    const float* pm = (const float*)d_in[16], *pv = (const float*)d_in[17];

    char* ws = (char*)d_ws;
    unsigned short* Wsw    = (unsigned short*)ws;                       // 3 * 512 KB
    float*          scales = (float*)(ws + (size_t)3 * C_ * C_ * 2);    // 6 * 512 f32
    unsigned char*  qs     = (unsigned char*)(ws + (size_t)2  * 1024 * 1024);
    unsigned char*  ks     = (unsigned char*)(ws + (size_t)18 * 1024 * 1024);
    unsigned char*  attn   = (unsigned char*)(ws + (size_t)34 * 1024 * 1024);

    prep_swizzle<<<192, 256, 0, stream>>>(q_w, k_w, proj_w, Wsw);
    prep_scales <<<6,   256, 0, stream>>>(qg, qb, qm, qv, kg, kb, km, kv,
                                          pg, pb, pm, pv, proj_b, scales);

    dim3 grid(N_ / 32, C_ / 128, B_);   // (32, 4, 8)
    gemm_bn_lif_f32<<<grid, 256, 0, stream>>>(q, Wsw,
                                              scales + 0,    scales + 512,  qs);
    gemm_bn_lif_f32<<<grid, 256, 0, stream>>>(k, Wsw + (size_t)C_ * C_,
                                              scales + 1024, scales + 1536, ks);
    qsum_attn<<<64, 256, 0, stream>>>(qs, attn);
    gemm_proj_bn_lif<<<grid, 256, 0, stream>>>(ks, attn,
                                               Wsw + (size_t)2 * C_ * C_,
                                               scales + 2048, scales + 2560,
                                               (float*)d_out);
}
